// NaiveSelfMix_72808285602172
// MI455X (gfx1250) — matrix operand order repaired and checked by reading, compiled, not run on hardware
//
#include <hip/hip_runtime.h>
#include <hip/hip_bf16.h>
#include <math.h>

// ---------------------------------------------------------------------------
// CDNA5 (gfx1250) equivariant block: pre-linear -> normgate -> CG tensor
// product ('uvu') -> post-linear -> normgate, residual sum.
// All channel contractions run on v_wmma_f32_16x16x32_bf16 (bf16 A/B, f32 acc).
// One wave (32 lanes) owns a 16-node tile; grid = N/16 blocks of 32 threads.
// ---------------------------------------------------------------------------

typedef __attribute__((ext_vector_type(16))) __bf16 bf16x16;
typedef __attribute__((ext_vector_type(8)))  float  f32x8;

#define FEAT   480
#define NMUL   224
#define TILE   16

// irrep table: muls {128,64,32}, dims {1,3,5}, offsets in merged layout
constexpr int MUL[3]  = {128, 64, 32};
constexpr int DIM[3]  = {1, 3, 5};
constexpr int BOFF[3] = {0, 128, 320};

// ---- CG paths ('uvu'); offsets into tp_w and into coupled channel axis ----
constexpr int P_I1[11]   = {0,0,0,1,1,1,1,2,2,2,2};
constexpr int P_I2[11]   = {0,1,2,0,1,1,2,0,1,2,2};
constexpr int P_L3[11]   = {0,1,2,1,0,2,1,2,1,0,2};
constexpr int P_WOFF[11] = {0,16384,24576,28672,36864,40960,45056,47104,51200,53248,54272};
constexpr int P_UOFF[11] = {0,0,0,128,128,128,192,192,256,192,224};
constexpr int P_E0[11]   = {0,1,4,9,12,15,26,37,42,53,58};
constexpr int P_EC[11]   = {1,3,5,3,3,11,11,5,11,5,25};

// ---- sparse real Wigner-3j tables (Frobenius-normalized) ----
struct W3E { int i, j, k; float v; };
#define R3 0.5773502692f
#define R5 0.4472135955f
#define WA 0.3162277660f   // 1/sqrt(10)
#define WC 0.1825741858f   // 1/sqrt(30)
#define W2C 0.3651483717f
#define S7 0.1195228609f   // 1/sqrt(70)
#define T7 0.2070196678f   // sqrt(3/70)
constexpr W3E C_W3J[83] = {
  // p0 (0,0,0)
  {0,0,0, 1.0f},
  // p1 (0,1,1)
  {0,0,0,R3},{0,1,1,R3},{0,2,2,R3},
  // p2 (0,2,2)
  {0,0,0,R5},{0,1,1,R5},{0,2,2,R5},{0,3,3,R5},{0,4,4,R5},
  // p3 (1,0,1)
  {0,0,0,R3},{1,0,1,R3},{2,0,2,R3},
  // p4 (1,1,0)
  {0,0,0,R3},{1,1,0,R3},{2,2,0,R3},
  // p5 (1,1,2)
  {2,0,0,WA},{0,2,0,WA},{0,1,1,WA},{1,0,1,WA},{1,1,2,W2C},{0,0,2,-WC},
  {2,2,2,-WC},{2,1,3,WA},{1,2,3,WA},{2,2,4,WA},{0,0,4,-WA},
  // p6 (1,2,1)  = p5 with (j,k) swapped
  {2,0,0,WA},{0,0,2,WA},{0,1,1,WA},{1,1,0,WA},{1,2,1,W2C},{0,2,0,-WC},
  {2,2,2,-WC},{2,3,1,WA},{1,3,2,WA},{2,4,2,WA},{0,4,0,-WA},
  // p7 (2,0,2)
  {0,0,0,R5},{1,0,1,R5},{2,0,2,R5},{3,0,3,R5},{4,0,4,R5},
  // p8 (2,1,1)  = p5 cyclic perm (k,i,j)
  {0,2,0,WA},{0,0,2,WA},{1,0,1,WA},{1,1,0,WA},{2,1,1,W2C},{2,0,0,-WC},
  {2,2,2,-WC},{3,2,1,WA},{3,1,2,WA},{4,2,2,WA},{4,0,0,-WA},
  // p9 (2,2,0)
  {0,0,0,R5},{1,1,0,R5},{2,2,0,R5},{3,3,0,R5},{4,4,0,R5},
  // p10 (2,2,2): fully symmetric, unit Frobenius norm
  {2,2,2, 2*S7},
  {0,0,2,-2*S7},{0,2,0,-2*S7},{2,0,0,-2*S7},
  {4,4,2,-2*S7},{4,2,4,-2*S7},{2,4,4,-2*S7},
  {1,1,2, S7},{1,2,1, S7},{2,1,1, S7},
  {3,3,2, S7},{3,2,3, S7},{2,3,3, S7},
  {0,1,3, T7},{0,3,1, T7},{1,0,3, T7},{1,3,0, T7},{3,0,1, T7},{3,1,0, T7},
  {1,1,4,-T7},{1,4,1,-T7},{4,1,1,-T7},
  {3,3,4, T7},{3,4,3, T7},{4,3,3, T7},
};

// ---------------------------------------------------------------------------
// WMMA fragment gathers. A: 16x32 (MxK), B: 32x16 (KxN), per CDNA5 ISA layout.
// a(m,k) = base[m*rs + k*cs]; b(k,n) = base[k*rs + n*cs].
// ---------------------------------------------------------------------------
__device__ __forceinline__ bf16x16 frag_a_f32(const float* base, int rs, int cs, int lane) {
  int m = lane & 15, h8 = (lane & 16) ? 8 : 0;
  bf16x16 f;
#pragma unroll
  for (int v = 0; v < 8; ++v) {
    int k0 = ((v & 4) ? 16 : 0) + h8 + 2 * (v & 3);
    f[2*v]   = (__bf16)base[m*rs + k0*cs];
    f[2*v+1] = (__bf16)base[m*rs + (k0+1)*cs];
  }
  return f;
}
__device__ __forceinline__ bf16x16 frag_a_bf16(const __bf16* base, int rs, int cs, int lane) {
  int m = lane & 15, h8 = (lane & 16) ? 8 : 0;
  bf16x16 f;
#pragma unroll
  for (int v = 0; v < 8; ++v) {
    int k0 = ((v & 4) ? 16 : 0) + h8 + 2 * (v & 3);
    f[2*v]   = base[m*rs + k0*cs];
    f[2*v+1] = base[m*rs + (k0+1)*cs];
  }
  return f;
}
__device__ __forceinline__ bf16x16 frag_b_f32(const float* base, int rs, int cs, int lane) {
  int n = lane & 15, kh = (lane & 16) ? 8 : 0;
  bf16x16 f;
#pragma unroll
  for (int v = 0; v < 8; ++v) {
    int k = ((v & 4) ? 16 : 0) + kh + 2 * (v & 3);
    f[2*v]   = (__bf16)base[k*rs + n*cs];
    f[2*v+1] = (__bf16)base[(k+1)*rs + n*cs];
  }
  return f;
}
__device__ __forceinline__ f32x8 wmma_bf16(bf16x16 a, bf16x16 b, f32x8 c) {
  return __builtin_amdgcn_wmma_f32_16x16x32_bf16(false, a, false, b, (short)0, c, false, false);
}
// fast silu: v_exp + v_rcp + mul (no IEEE divide expansion)
__device__ __forceinline__ float silu(float x) {
  return x * __builtin_amdgcn_rcpf(1.f + __expf(-x));
}

// ---------------------------------------------------------------------------
// NormGate: invariants -> layernorm(224) -> 2x WMMA MLP with silu -> gate bf16
// s_x: [16][480] f32 merged blocks (LDS). s_vb/s_h/s_g: bf16 [16*224] (LDS).
// ---------------------------------------------------------------------------
__device__ __forceinline__ void normgate(const float* s_x,
                                         const float* w1, const float* b1,
                                         const float* w2, const float* b2,
                                         __bf16* s_vb, __bf16* s_h, __bf16* s_g,
                                         int lane) {
  if (lane < TILE) {
    const int n = lane;
    float m = 0.f, s2 = 0.f;
    for (int c = 0; c < 128; ++c) {
      float val = s_x[n*FEAT + c];
      s_vb[n*NMUL + c] = (__bf16)val; m += val; s2 += val*val;
    }
    for (int c = 0; c < 64; ++c) {
      const float* p = s_x + n*FEAT + 128 + c*3;
      float val = sqrtf(p[0]*p[0] + p[1]*p[1] + p[2]*p[2] + 1e-12f);
      s_vb[n*NMUL + 128 + c] = (__bf16)val; m += val; s2 += val*val;
    }
    for (int c = 0; c < 32; ++c) {
      const float* p = s_x + n*FEAT + 320 + c*5;
      float val = sqrtf(p[0]*p[0] + p[1]*p[1] + p[2]*p[2] + p[3]*p[3] + p[4]*p[4] + 1e-12f);
      s_vb[n*NMUL + 192 + c] = (__bf16)val; m += val; s2 += val*val;
    }
    m *= (1.f / NMUL);
    float inv = rsqrtf(s2 * (1.f / NMUL) - m*m + 1e-5f);
    for (int c = 0; c < NMUL; ++c)
      s_vb[n*NMUL + c] = (__bf16)(((float)s_vb[n*NMUL + c] - m) * inv);
  }
  __syncthreads();
  int ncol = lane & 15, mh = (lane & 16) ? 8 : 0;
  // GEMM1: [16,224] x [224,224], silu(.+b1) -> s_h (bf16)
  for (int nt = 0; nt < 14; ++nt) {
    f32x8 acc = {};
#pragma unroll
    for (int kt = 0; kt < 7; ++kt) {
      bf16x16 af = frag_a_bf16(s_vb + kt*32, NMUL, 1, lane);
      bf16x16 bf = frag_b_f32(w1 + kt*32*NMUL + nt*16, NMUL, 1, lane);
      acc = wmma_bf16(af, bf, acc);
    }
    int col = nt*16 + ncol;
    float bias = b1[col];
#pragma unroll
    for (int r = 0; r < 8; ++r)
      s_h[(r + mh)*NMUL + col] = (__bf16)silu(acc[r] + bias);
  }
  __syncthreads();
  // GEMM2: [16,224] x [224,224], silu(.+b2) -> s_g (bf16)
  for (int nt = 0; nt < 14; ++nt) {
    f32x8 acc = {};
#pragma unroll
    for (int kt = 0; kt < 7; ++kt) {
      bf16x16 af = frag_a_bf16(s_h + kt*32, NMUL, 1, lane);
      bf16x16 bf = frag_b_f32(w2 + kt*32*NMUL + nt*16, NMUL, 1, lane);
      acc = wmma_bf16(af, bf, acc);
    }
    int col = nt*16 + ncol;
    float bias = b2[col];
#pragma unroll
    for (int r = 0; r < 8; ++r)
      s_g[(r + mh)*NMUL + col] = (__bf16)silu(acc[r] + bias);
  }
  __syncthreads();
}

// ---------------------------------------------------------------------------
// Kernel 1: pre-linear + normgate; writes gated pre into d_out (temp).
// ---------------------------------------------------------------------------
__global__ __launch_bounds__(32)
void k_pre(const float* __restrict__ feat,
           const float* __restrict__ w0, const float* __restrict__ w1l, const float* __restrict__ w2l,
           const float* __restrict__ ngw1, const float* __restrict__ ngb1,
           const float* __restrict__ ngw2, const float* __restrict__ ngb2,
           float* __restrict__ out) {
  __shared__ float  s_pre[TILE*FEAT];   // 30720 B
  __shared__ __bf16 s_vb[TILE*NMUL];    //  7168 B
  __shared__ __bf16 s_h [TILE*NMUL];    //  7168 B
  __shared__ __bf16 s_g [TILE*NMUL];    //  7168 B

  int lane = threadIdx.x;
  long node0 = (long)blockIdx.x * TILE;
  int ncol = lane & 15, mh = (lane & 16) ? 8 : 0;

  // l=0: [16,128] x [128,128] / sqrt(128)
#pragma unroll
  for (int nt = 0; nt < 8; ++nt) {
    f32x8 acc = {};
#pragma unroll
    for (int kt = 0; kt < 4; ++kt) {
      bf16x16 af = frag_a_f32(feat + node0*FEAT + kt*32, FEAT, 1, lane);
      bf16x16 bf = frag_b_f32(w0 + kt*32*128 + nt*16, 128, 1, lane);
      acc = wmma_bf16(af, bf, acc);
    }
#pragma unroll
    for (int r = 0; r < 8; ++r)
      s_pre[(r + mh)*FEAT + nt*16 + ncol] = acc[r] * 0.0883883476f;
  }
  // l=1: per j, [16,64] x [64,64] / 8
#pragma unroll
  for (int j = 0; j < 3; ++j) {
#pragma unroll
    for (int nt = 0; nt < 4; ++nt) {
      f32x8 acc = {};
#pragma unroll
      for (int kt = 0; kt < 2; ++kt) {
        bf16x16 af = frag_a_f32(feat + node0*FEAT + 128 + j + kt*32*3, FEAT, 3, lane);
        bf16x16 bf = frag_b_f32(w1l + kt*32*64 + nt*16, 64, 1, lane);
        acc = wmma_bf16(af, bf, acc);
      }
#pragma unroll
      for (int r = 0; r < 8; ++r)
        s_pre[(r + mh)*FEAT + 128 + (nt*16 + ncol)*3 + j] = acc[r] * 0.125f;
    }
  }
  // l=2: per j, [16,32] x [32,32] / sqrt(32)
#pragma unroll
  for (int j = 0; j < 5; ++j) {
#pragma unroll
    for (int nt = 0; nt < 2; ++nt) {
      bf16x16 af = frag_a_f32(feat + node0*FEAT + 320 + j, FEAT, 5, lane);
      bf16x16 bf = frag_b_f32(w2l + nt*16, 32, 1, lane);
      f32x8 acc = {};
      acc = wmma_bf16(af, bf, acc);
#pragma unroll
      for (int r = 0; r < 8; ++r)
        s_pre[(r + mh)*FEAT + 320 + (nt*16 + ncol)*5 + j] = acc[r] * 0.1767766953f;
    }
  }
  __syncthreads();

  normgate(s_pre, ngw1, ngb1, ngw2, ngb2, s_vb, s_h, s_g, lane);

  for (int idx = lane; idx < TILE * FEAT; idx += 32) {
    int n = idx / FEAT, c = idx - n * FEAT;
    int u = (c < 128) ? c : (c < 320) ? 128 + (c - 128) / 3 : 192 + (c - 320) / 5;
    out[(node0 + n) * FEAT + c] = s_pre[idx] * (float)s_g[n*NMUL + u];
  }
}

// ---------------------------------------------------------------------------
// One CG path, fully compile-time specialized: t-mix GEMM -> w3j elementwise
// -> post-linear GEMM fused into LDS accumulator. 32-channel u-chunks.
// ---------------------------------------------------------------------------
template<int P>
__device__ __forceinline__ void path_body(const float* __restrict__ feat,
                                          const float* __restrict__ tpw,
                                          const float* __restrict__ pw0,
                                          const float* __restrict__ pw1,
                                          const float* __restrict__ pw2,
                                          const float* __restrict__ preg,
                                          long node0, int lane,
                                          __bf16* s_t, __bf16* s_y, float* s_post) {
  constexpr int I1 = P_I1[P], I2 = P_I2[P], L3 = P_L3[P];
  constexpr int M1 = MUL[I1], M2 = MUL[I2];
  constexpr int D1 = DIM[I1], D2 = DIM[I2], D3 = DIM[L3];
  constexpr int OFF1 = BOFF[I1], OFF2 = BOFF[I2], OO = BOFF[L3];
  constexpr int NOUT = MUL[L3];
  constexpr int WOFF = P_WOFF[P], UOFF = P_UOFF[P];
  constexpr int E0 = P_E0[P], EC = P_EC[P];
  constexpr float INV_M2 = (M2 == 128) ? 0.0883883476f : (M2 == 64) ? 0.125f : 0.1767766953f;
  const float* pw = (L3 == 0) ? pw0 : (L3 == 1) ? pw1 : pw2;
  int ncol = lane & 15, mh = (lane & 16) ? 8 : 0;

  for (int uc = 0; uc < M1 / 32; ++uc) {
    int u0 = uc * 32;
    // ---- t = (pre_gated mixed by tp_w)/sqrt(M2): rows (n,j), K=M2, N=32
#pragma unroll
    for (int j = 0; j < D2; ++j) {
#pragma unroll
      for (int nt = 0; nt < 2; ++nt) {
        f32x8 acc = {};
#pragma unroll
        for (int kt = 0; kt < M2 / 32; ++kt) {
          bf16x16 af = frag_a_f32(preg + node0*FEAT + OFF2 + j + kt*32*D2, FEAT, D2, lane);
          bf16x16 bf = frag_b_f32(tpw + WOFF + (u0 + nt*16)*M2 + kt*32, 1, M2, lane);
          acc = wmma_bf16(af, bf, acc);
        }
#pragma unroll
        for (int r = 0; r < 8; ++r)
          s_t[(r + mh)*32*D2 + (nt*16 + ncol)*D2 + j] = (__bf16)(acc[r] * INV_M2);
      }
    }
    __syncthreads();
    // ---- y[n,u,k] = sum_{i,j} b1[n,u,i]*t[n,u,j]*w3j[i,j,k]  (registers only)
#pragma unroll
    for (int it = 0; it < 16; ++it) {
      int idx = it*32 + lane;
      int n = idx & 15, uu = idx >> 4;
      float b1v[D1], tv[D2], yv[D3];
#pragma unroll
      for (int i = 0; i < D1; ++i)
        b1v[i] = feat[(node0 + n)*FEAT + OFF1 + (u0 + uu)*D1 + i];
#pragma unroll
      for (int j = 0; j < D2; ++j)
        tv[j] = (float)s_t[n*32*D2 + uu*D2 + j];
#pragma unroll
      for (int k = 0; k < D3; ++k) yv[k] = 0.f;
#pragma unroll
      for (int e = E0; e < E0 + EC; ++e)
        yv[C_W3J[e].k] += b1v[C_W3J[e].i] * tv[C_W3J[e].j] * C_W3J[e].v;
#pragma unroll
      for (int k = 0; k < D3; ++k)
        s_y[n*32*D3 + uu*D3 + k] = (__bf16)yv[k];
    }
    __syncthreads();
    // ---- post-linear slice: rows (n,k), K=32, N=NOUT; accumulate in s_post
#pragma unroll
    for (int k = 0; k < D3; ++k) {
      bf16x16 af = frag_a_bf16(s_y + k, 32*D3, D3, lane);
#pragma unroll
      for (int nt = 0; nt < NOUT / 16; ++nt) {
        bf16x16 bf = frag_b_f32(pw + (UOFF + u0)*NOUT + nt*16, NOUT, 1, lane);
        f32x8 acc = {};
        acc = wmma_bf16(af, bf, acc);
#pragma unroll
        for (int r = 0; r < 8; ++r)
          s_post[(r + mh)*FEAT + OO + (nt*16 + ncol)*D3 + k] += acc[r];
      }
    }
    __syncthreads();
  }
}

// ---------------------------------------------------------------------------
// Kernel 2: CG couple + post-linear + post normgate + residual sum.
// Reads gated pre from d_out, overwrites d_out with the final result.
// ---------------------------------------------------------------------------
__global__ __launch_bounds__(32)
void k_cgpost(const float* __restrict__ feat, const float* __restrict__ tpw,
              const float* __restrict__ pw0, const float* __restrict__ pw1,
              const float* __restrict__ pw2,
              const float* __restrict__ ngw1, const float* __restrict__ ngb1,
              const float* __restrict__ ngw2, const float* __restrict__ ngb2,
              float* __restrict__ out) {
  __shared__ float  s_post[TILE*FEAT];   // 30720 B
  __shared__ __bf16 s_t[TILE*32*5];      //  5120 B
  __shared__ __bf16 s_y[TILE*32*5];      //  5120 B
  __shared__ __bf16 s_vb[TILE*NMUL];     //  7168 B
  __shared__ __bf16 s_h [TILE*NMUL];     //  7168 B
  __shared__ __bf16 s_g [TILE*NMUL];     //  7168 B

  int lane = threadIdx.x;
  long node0 = (long)blockIdx.x * TILE;

  for (int idx = lane; idx < TILE * FEAT; idx += 32) s_post[idx] = 0.f;
  __syncthreads();

  path_body<0 >(feat, tpw, pw0, pw1, pw2, out, node0, lane, s_t, s_y, s_post);
  path_body<1 >(feat, tpw, pw0, pw1, pw2, out, node0, lane, s_t, s_y, s_post);
  path_body<2 >(feat, tpw, pw0, pw1, pw2, out, node0, lane, s_t, s_y, s_post);
  path_body<3 >(feat, tpw, pw0, pw1, pw2, out, node0, lane, s_t, s_y, s_post);
  path_body<4 >(feat, tpw, pw0, pw1, pw2, out, node0, lane, s_t, s_y, s_post);
  path_body<5 >(feat, tpw, pw0, pw1, pw2, out, node0, lane, s_t, s_y, s_post);
  path_body<6 >(feat, tpw, pw0, pw1, pw2, out, node0, lane, s_t, s_y, s_post);
  path_body<7 >(feat, tpw, pw0, pw1, pw2, out, node0, lane, s_t, s_y, s_post);
  path_body<8 >(feat, tpw, pw0, pw1, pw2, out, node0, lane, s_t, s_y, s_post);
  path_body<9 >(feat, tpw, pw0, pw1, pw2, out, node0, lane, s_t, s_y, s_post);
  path_body<10>(feat, tpw, pw0, pw1, pw2, out, node0, lane, s_t, s_y, s_post);

  // ---- /sqrt(coupled K) per output irrep: 1/sqrt(224, 288, 256)
  for (int idx = lane; idx < TILE * FEAT; idx += 32) {
    int c = idx % FEAT;
    s_post[idx] *= (c < 128) ? 0.066815310f : (c < 320) ? 0.058925565f : 0.0625f;
  }
  __syncthreads();

  normgate(s_post, ngw1, ngb1, ngw2, ngb2, s_vb, s_h, s_g, lane);

  // ---- out = feat + pre(gated, currently in out) + post(gated)
  for (int idx = lane; idx < TILE * FEAT; idx += 32) {
    int n = idx / FEAT, c = idx - n * FEAT;
    int u = (c < 128) ? c : (c < 320) ? 128 + (c - 128) / 3 : 192 + (c - 320) / 5;
    long g = (node0 + n) * FEAT + c;
    float pre = out[g];
    out[g] = feat[g] + pre + s_post[idx] * (float)s_g[n*NMUL + u];
  }
}

// ---------------------------------------------------------------------------
extern "C" void kernel_launch(void* const* d_in, const int* in_sizes, int n_in,
                              void* d_out, int out_size, void* d_ws, size_t ws_size,
                              hipStream_t stream) {
  const float* feat    = (const float*)d_in[0];
  const float* pre_w0  = (const float*)d_in[1];
  const float* pre_w1  = (const float*)d_in[2];
  const float* pre_w2  = (const float*)d_in[3];
  const float* png_w1  = (const float*)d_in[4];
  const float* png_b1  = (const float*)d_in[5];
  const float* png_w2  = (const float*)d_in[6];
  const float* png_b2  = (const float*)d_in[7];
  const float* tp_w    = (const float*)d_in[8];
  const float* post_w0 = (const float*)d_in[9];
  const float* post_w1 = (const float*)d_in[10];
  const float* post_w2 = (const float*)d_in[11];
  const float* qng_w1  = (const float*)d_in[12];
  const float* qng_b1  = (const float*)d_in[13];
  const float* qng_w2  = (const float*)d_in[14];
  const float* qng_b2  = (const float*)d_in[15];
  float* out = (float*)d_out;

  int n_nodes = in_sizes[0] / FEAT;
  dim3 grid(n_nodes / TILE), block(32);
  k_pre<<<grid, block, 0, stream>>>(feat, pre_w0, pre_w1, pre_w2,
                                    png_w1, png_b1, png_w2, png_b2, out);
  k_cgpost<<<grid, block, 0, stream>>>(feat, tp_w, post_w0, post_w1, post_w2,
                                       qng_w1, qng_b1, qng_w2, qng_b2, out);
}
